// BMM_S8T_S8N_S32T_WITH_SCALING_74182675137262
// MI455X (gfx1250) — 8-bit matrix operand order repaired and checked by reading, compiled, not run on hardware
//
#include <hip/hip_runtime.h>
#include <cstdint>
#include <cstddef>


typedef __attribute__((ext_vector_type(8))) int v8i;
typedef __attribute__((__vector_size__(16))) int v4i32;

#define TILE 128
#define LDS_STRIDE 144

#if defined(__AMDGCN__) && __has_builtin(__builtin_amdgcn_global_load_async_to_lds_b128) && \
    __has_builtin(__builtin_amdgcn_s_wait_asynccnt)
#define USE_ASYNC_LDS 1
#endif

__global__ __launch_bounds__(256)
void BMM_S8T_S8N_S32T_WITH_SCALING_kernel(const int8_t* __restrict__ A,
                                          const int8_t* __restrict__ Bm,
                                          const float* __restrict__ alphaPtr,
                                          int* __restrict__ Out,
                                          int M, int N, int K)
{
    __shared__ unsigned char ldsB[TILE * LDS_STRIDE];

    const int b  = blockIdx.z;
    const int tm = blockIdx.y * TILE;
    const int tn = blockIdx.x * TILE;

    const int tid  = threadIdx.x;
    const int lane = tid & 31;
    const int wave = tid >> 5;

    const float alpha = *alphaPtr;

    {
        const int8_t* bTile = Bm + ((size_t)b * N + tn) * K;
        #pragma unroll
        for (int i = 0; i < 4; ++i) {
            const int c   = tid + i * 256;
            const int row = c >> 3;
            const int off = (c & 7) << 4;
#ifdef USE_ASYNC_LDS
            __builtin_amdgcn_global_load_async_to_lds_b128(
                (__attribute__((address_space(1))) v4i32*)(bTile + (size_t)row * K + off),
                (__attribute__((address_space(3))) v4i32*)&ldsB[row * LDS_STRIDE + off],
                 0,  0);
#else
            const uint4 v = *reinterpret_cast<const uint4*>(bTile + (size_t)row * K + off);
            *reinterpret_cast<uint4*>(&ldsB[row * LDS_STRIDE + off]) = v;
#endif
        }
    }

    const int rowA  = lane & 15;
    const int koff8 = (lane >> 4) << 3;
    const int8_t* aRow = A + ((size_t)b * M + tm + wave * 16 + rowA) * K;
    __builtin_prefetch(aRow, 0, 1);

#ifdef USE_ASYNC_LDS
    __builtin_amdgcn_s_wait_asynccnt(0);
#endif
    __syncthreads();

    v8i afrag[2];
    #pragma unroll
    for (int h = 0; h < 2; ++h) {
        #pragma unroll
        for (int t = 0; t < 4; ++t) {
            const uint2 u = *reinterpret_cast<const uint2*>(aRow + h * 64 + (t & 1) * 8 + (t >> 1) * 32 + 2 * koff8);
            afrag[h][2 * t]     = (int)u.x;
            afrag[h][2 * t + 1] = (int)u.y;
        }
    }

    v8i acc[8] = {};

    const int colB  = lane & 15;
    const int klane = (lane >> 4) << 4;

    #pragma unroll
    for (int nt = 0; nt < 8; ++nt) {
        #pragma unroll
        for (int h = 0; h < 2; ++h) {
            const unsigned char* p = &ldsB[(nt * 16 + colB) * LDS_STRIDE + h * 64 + klane];
            const uint4 lo = *reinterpret_cast<const uint4*>(p);
            const uint4 hi = *reinterpret_cast<const uint4*>(p + 32);
            v8i bfrag;
            bfrag[0] = (int)lo.x; bfrag[1] = (int)lo.y;
            bfrag[2] = (int)lo.z; bfrag[3] = (int)lo.w;
            bfrag[4] = (int)hi.x; bfrag[5] = (int)hi.y;
            bfrag[6] = (int)hi.z; bfrag[7] = (int)hi.w;
            acc[nt] = __builtin_amdgcn_wmma_i32_16x16x64_iu8(
                true, afrag[h], true, bfrag, acc[nt], false, false);
        }
    }

    const int mBase = tm + wave * 16 + ((lane >> 4) << 3);
    const int nLane = tn + (lane & 15);
    #pragma unroll
    for (int nt = 0; nt < 8; ++nt) {
        #pragma unroll
        for (int r = 0; r < 8; ++r) {
            const int   m = mBase + r;
            const int   n = nLane + nt * 16;
            const float f = (float)acc[nt][r] * alpha;
            Out[((size_t)b * M + m) * (size_t)N + n] = (int)__builtin_rintf(f);
        }
    }
}

extern "C" void kernel_launch(void* const* d_in, const int* in_sizes, int n_in,
                              void* d_out, int out_size, void* d_ws, size_t ws_size,
                              hipStream_t stream) {
    (void)in_sizes; (void)n_in; (void)out_size; (void)d_ws; (void)ws_size;

    const int8_t* A     = (const int8_t*)d_in[0];
    const int8_t* Bmat  = (const int8_t*)d_in[1];
    const float*  alpha = (const float*)d_in[2];
    int*          out   = (int*)d_out;

    const int Bn = 64, M = 1024, N = 1024, K = 128;
    dim3 grid(N / TILE, M / TILE, Bn);
    dim3 block(256);
    BMM_S8T_S8N_S32T_WITH_SCALING_kernel<<<grid, block, 0, stream>>>(A, Bmat, alpha, out, M, N, K);
}
